// KLCLR_89146341196337
// MI455X (gfx1250) — matrix operand order repaired and checked by reading, not compiled, not run on hardware
//
#include <hip/hip_runtime.h>

// ---------------------------------------------------------------------------
// Problem: N=16384, Z=128, K=512 (x2 codebooks -> 1024 rows), D_IN=10000
// Output: concat(z_e [N*Z], z_c [N*Z]) fp32
//
// Collapsed algorithm (argmax of normalized monotone kernel == argmin dist;
// MLP applied to only the 1024 distinct codebook rows, then gathered):
//   prep :  Eh = f16(concat(e1,e2)), enorm[k] = ||e_k||^2
//   wprep:  W1T = f16(W1^T) [10112x128], W2T = f16(W2^T) [128x10112] (0-pad)
//   mlp  :  C[1024x128] = relu(Ecat@W1+b1)@W2 + b2
//           (chunked over D_IN, double-buffered GLOBAL_LOAD_ASYNC_TO_LDS)
//   assign: WMMA scores, k = argmin(enorm - 2*dot), subject select,
//           out_c = gather C, out_z = z_e copy
// ---------------------------------------------------------------------------

typedef __attribute__((ext_vector_type(16))) _Float16 v16h;
typedef __attribute__((ext_vector_type(8)))  _Float16 v8h;
typedef __attribute__((ext_vector_type(8)))  float    v8f;

#define DIN    10000
#define DINP   10112          // 79 * 128, zero-padded
#define NCHUNK 79
#define ZD     128
#define KCAT   1024
#define NROW   16384

__device__ __forceinline__ v8f vzero8() {
  v8f z;
  #pragma unroll
  for (int i = 0; i < 8; ++i) z[i] = 0.0f;
  return z;
}

__device__ __forceinline__ v8f wmma_f16(v16h a, v16h b, v8f c) {
  return __builtin_amdgcn_wmma_f32_16x16x32_f16(
      false, a, false, b, (short)0, c, false, false);
}

// --- CDNA5 async copy: 16 bytes global -> LDS per lane (ASYNCcnt) ----------
__device__ __forceinline__ void async_copy_b128(unsigned lds_addr,
                                                const void* gsrc) {
  unsigned long long ga = (unsigned long long)gsrc;
  asm volatile("global_load_async_to_lds_b128 %0, %1, off"
               :: "v"(lds_addr), "v"(ga) : "memory");
}
__device__ __forceinline__ void wait_async0() {
  asm volatile("s_wait_asynccnt 0x0" ::: "memory");
}
__device__ __forceinline__ unsigned lds_addr_of(const void* p) {
  return (unsigned)(unsigned long long)p;   // flat LDS aperture: addr[31:0]
}

// A-fragment (16x32 f16, row-major src [M][K]); m = m0+lane%16.
__device__ __forceinline__ v16h frag_a(const _Float16* src, int ld, int m,
                                       int k0, int halfsel) {
  const v8h lo = *(const v8h*)(src + m * ld + k0 + 8 * halfsel);
  const v8h hi = *(const v8h*)(src + m * ld + k0 + 16 + 8 * halfsel);
  v16h a;
  #pragma unroll
  for (int i = 0; i < 8; ++i) { a[i] = lo[i]; a[8 + i] = hi[i]; }
  return a;
}
// B-fragment (32x16 f16) from TRANSPOSED store [N][K].
__device__ __forceinline__ v16h frag_bt(const _Float16* srcT, int ld, int n,
                                        int k0, int halfsel) {
  const v8h lo = *(const v8h*)(srcT + n * ld + k0 + 8 * halfsel);
  const v8h hi = *(const v8h*)(srcT + n * ld + k0 + 16 + 8 * halfsel);
  v16h b;
  #pragma unroll
  for (int i = 0; i < 8; ++i) { b[i] = lo[i]; b[8 + i] = hi[i]; }
  return b;
}

__device__ __forceinline__ void argmin_reduce16(float& v, int& i) {
  #pragma unroll
  for (int off = 8; off >= 1; off >>= 1) {
    float ov = __shfl_xor(v, off, 32);
    int   oi = __shfl_xor(i, off, 32);
    if (ov < v || (ov == v && oi < i)) { v = ov; i = oi; }
  }
}

// ---------------------------------------------------------------------------
// Kernel 1: codebooks -> f16 + row norms. 1 wave per row.
// ---------------------------------------------------------------------------
__global__ __launch_bounds__(256) void KLCLR_prep(const float* __restrict__ e1,
                                                  const float* __restrict__ e2,
                                                  _Float16* __restrict__ Eh,
                                                  float* __restrict__ enorm) {
  int gwave = (blockIdx.x * blockDim.x + threadIdx.x) >> 5;  // 0..1023
  int lane  = threadIdx.x & 31;
  if (gwave >= KCAT) return;
  const float* src = (gwave < 512) ? (e1 + (size_t)gwave * ZD)
                                   : (e2 + (size_t)(gwave - 512) * ZD);
  float s = 0.0f;
  #pragma unroll
  for (int i = 0; i < 4; ++i) {
    float v = src[lane * 4 + i];
    s += v * v;
    Eh[(size_t)gwave * ZD + lane * 4 + i] = (_Float16)v;
  }
  #pragma unroll
  for (int off = 16; off >= 1; off >>= 1) s += __shfl_xor(s, off, 32);
  if (lane == 0) enorm[gwave] = s;
}

// ---------------------------------------------------------------------------
// Kernel 1b: weights -> f16 transposed + zero-padded to DINP.
//   W1T[d][k] = W1[k][d]  (d<DIN else 0)   [DINP x 128]
//   W2T[n][d] = W2[d][n]  (d<DIN else 0)   [128 x DINP]
// Coalesced writes; scattered reads hit L2 once.
// ---------------------------------------------------------------------------
__global__ __launch_bounds__(256) void KLCLR_wprep(const float* __restrict__ W1,
                                                   const float* __restrict__ W2,
                                                   _Float16* __restrict__ W1T,
                                                   _Float16* __restrict__ W2T) {
  const size_t total = (size_t)DINP * ZD;   // 1,294,336
  const size_t stride = (size_t)gridDim.x * 256;
  for (size_t idx = (size_t)blockIdx.x * 256 + threadIdx.x; idx < total;
       idx += stride) {
    // W1T flat: idx = d*128 + k
    {
      int d = (int)(idx >> 7), k = (int)(idx & 127);
      float v = (d < DIN) ? W1[(size_t)k * DIN + d] : 0.0f;
      W1T[idx] = (_Float16)v;
    }
    // W2T flat: idx = n*DINP + d
    {
      int n = (int)(idx / DINP), d = (int)(idx % DINP);
      float v = (d < DIN) ? W2[(size_t)d * ZD + n] : 0.0f;
      W2T[idx] = (_Float16)v;
    }
  }
}

// Stage one 128x128-half chunk (32KB) into LDS via per-lane async B128 copies.
// dst row r (256B) <- src row base + r*rowStrideHalfs, 16B segments.
__device__ __forceinline__ void stage_async(_Float16* dstLds,
                                            const _Float16* srcBase,
                                            size_t rowStrideHalfs, int tid) {
  #pragma unroll
  for (int i = 0; i < 8; ++i) {
    int idx = tid + 256 * i;            // 0..2047
    int row = idx >> 4, seg = idx & 15; // 128 rows x 16 segs
    const void* src =
        (const char*)(srcBase + (size_t)row * rowStrideHalfs) + seg * 16;
    unsigned dst = lds_addr_of(dstLds + row * 128) + seg * 16;
    async_copy_b128(dst, src);
  }
}

// ---------------------------------------------------------------------------
// Kernel 2: fused MLP over 1024 codebook rows, async double-buffered weights.
// Block: 256 thr (8 waves), 64 rows of Ecat.
// Dynamic LDS 96KB: Esh 16K | Hsh 16K | Wbuf0 32K (W1 chunk) | Wbuf1 32K (W2).
// ---------------------------------------------------------------------------
__global__ __launch_bounds__(256) void KLCLR_mlp(const float* __restrict__ e1,
                                                 const float* __restrict__ e2,
                                                 const _Float16* __restrict__ W1T,
                                                 const _Float16* __restrict__ W2T,
                                                 const float* __restrict__ b1,
                                                 const float* __restrict__ b2,
                                                 float* __restrict__ Cws) {
  extern __shared__ char smem[];
  _Float16* Esh   = (_Float16*)(smem);              // 64*128   = 16KB
  _Float16* Hsh   = (_Float16*)(smem + 16 * 1024);  // 64*128   = 16KB
  _Float16* Wbuf0 = (_Float16*)(smem + 32 * 1024);  // 128*128  = 32KB
  _Float16* Wbuf1 = (_Float16*)(smem + 64 * 1024);  // 128*128  = 32KB

  const int tid  = threadIdx.x;
  const int base = blockIdx.x * 64;

  // Prologue: async-stage W1 chunk 0, VGPR-stage E tile (needs f32->f16)
  stage_async(Wbuf0, W1T, 128, tid);
  for (int idx = tid; idx < 64 * 128; idx += 256) {
    int m = idx >> 7, k = idx & 127;
    int row = base + m;
    float v = (row < 512) ? e1[(size_t)row * ZD + k]
                          : e2[(size_t)(row - 512) * ZD + k];
    Esh[idx] = (_Float16)v;
  }
  wait_async0();
  __syncthreads();

  const int wave = tid >> 5, lane = tid & 31;
  const int halfsel = lane >> 4, ln = lane & 15;
  const int m0  = (wave & 3) * 16;
  const int ntb = (wave >> 2) * 4;

  v16h ae[4];
  #pragma unroll
  for (int ks = 0; ks < 4; ++ks) ae[ks] = frag_a(Esh, 128, m0 + ln, 32 * ks, halfsel);

  v8f Cacc[4];
  #pragma unroll
  for (int j = 0; j < 4; ++j) Cacc[j] = vzero8();

  for (int c = 0; c < NCHUNK; ++c) {
    const int d0 = c * 128;

    // Overlap: pull W2 chunk into buf1 while GEMM1 runs out of buf0
    stage_async(Wbuf1, W2T + d0, DINP, tid);

    // H = relu(E @ W1chunk + b1chunk)
    #pragma unroll
    for (int j = 0; j < 4; ++j) {
      const int n0 = (ntb + j) * 16;
      v8f acc = vzero8();
      #pragma unroll
      for (int ks = 0; ks < 4; ++ks)
        acc = wmma_f16(ae[ks], frag_bt(Wbuf0, 128, n0 + ln, 32 * ks, halfsel), acc);
      const int dg = d0 + n0 + ln;
      const float bias = (dg < DIN) ? b1[dg] : 0.0f;
      #pragma unroll
      for (int r = 0; r < 8; ++r) {
        float h = acc[r] + bias;
        h = h > 0.0f ? h : 0.0f;
        Hsh[(m0 + r + 8 * halfsel) * 128 + n0 + ln] = (_Float16)h;
      }
    }
    wait_async0();        // W2 chunk landed
    __syncthreads();      // Hsh complete, buf0 free, buf1 visible

    // Overlap: pull next W1 chunk into buf0 while GEMM2 runs out of buf1
    if (c + 1 < NCHUNK)
      stage_async(Wbuf0, W1T + (size_t)(d0 + 128) * 128, 128, tid);

    // C += H @ W2chunk
    v16h ah[4];
    #pragma unroll
    for (int ks = 0; ks < 4; ++ks) ah[ks] = frag_a(Hsh, 128, m0 + ln, 32 * ks, halfsel);
    #pragma unroll
    for (int j = 0; j < 4; ++j) {
      const int n0 = (ntb + j) * 16;
      #pragma unroll
      for (int ks = 0; ks < 4; ++ks)
        Cacc[j] = wmma_f16(ah[ks], frag_bt(Wbuf1, 128, n0 + ln, 32 * ks, halfsel), Cacc[j]);
    }
    wait_async0();        // next W1 chunk landed
    __syncthreads();      // buf1/Hsh free for next iteration
  }

  // Epilogue: + b2, write C rows (f32)
  #pragma unroll
  for (int j = 0; j < 4; ++j) {
    const int n  = (ntb + j) * 16 + ln;
    const float bb = b2[n];
    #pragma unroll
    for (int r = 0; r < 8; ++r)
      Cws[(size_t)(base + m0 + r + 8 * halfsel) * ZD + n] = Cacc[j][r] + bb;
  }
}

// ---------------------------------------------------------------------------
// Kernel 3: assignment + subject select + gather + z_e passthrough.
// ---------------------------------------------------------------------------
__global__ __launch_bounds__(256) void KLCLR_assign(
    const float* __restrict__ z, const int* __restrict__ subject,
    const _Float16* __restrict__ Eh, const float* __restrict__ enorm,
    const float* __restrict__ Cws, float* __restrict__ out_z,
    float* __restrict__ out_c) {
  __shared__ _Float16 zsh[128 * 128];
  __shared__ int codes[128];

  const int tid  = threadIdx.x;
  const size_t base = (size_t)blockIdx.x * 128;

  for (int idx = tid; idx < 128 * 128; idx += 256) {
    float v = z[base * ZD + idx];
    out_z[base * ZD + idx] = v;
    zsh[idx] = (_Float16)v;
  }
  __syncthreads();

  const int wave = tid >> 5, lane = tid & 31;
  const int halfsel = lane >> 4, ln = lane & 15;
  const int m0 = wave * 16;

  v16h a[4];
  #pragma unroll
  for (int ks = 0; ks < 4; ++ks) a[ks] = frag_a(zsh, 128, m0 + ln, 32 * ks, halfsel);

  float bv1[8], bv2[8];
  int   bi1[8], bi2[8];
  #pragma unroll
  for (int r = 0; r < 8; ++r) {
    bv1[r] = 3.402823466e38f; bv2[r] = 3.402823466e38f;
    bi1[r] = 0;               bi2[r] = 512;
  }

  for (int nt = 0; nt < 32; ++nt) {          // codebook 1
    const int n = nt * 16 + ln;
    v8f acc = vzero8();
    #pragma unroll
    for (int ks = 0; ks < 4; ++ks)
      acc = wmma_f16(a[ks], frag_bt(Eh, 128, n, 32 * ks, halfsel), acc);
    const float en = enorm[n];
    #pragma unroll
    for (int r = 0; r < 8; ++r) {
      float val = en - 2.0f * acc[r];
      if (val < bv1[r]) { bv1[r] = val; bi1[r] = n; }
    }
  }
  for (int nt = 32; nt < 64; ++nt) {         // codebook 2
    const int n = nt * 16 + ln;
    v8f acc = vzero8();
    #pragma unroll
    for (int ks = 0; ks < 4; ++ks)
      acc = wmma_f16(a[ks], frag_bt(Eh, 128, n, 32 * ks, halfsel), acc);
    const float en = enorm[n];
    #pragma unroll
    for (int r = 0; r < 8; ++r) {
      float val = en - 2.0f * acc[r];
      if (val < bv2[r]) { bv2[r] = val; bi2[r] = n; }
    }
  }

  #pragma unroll
  for (int r = 0; r < 8; ++r) {
    argmin_reduce16(bv1[r], bi1[r]);
    argmin_reduce16(bv2[r], bi2[r]);
  }

  if (ln == 0) {
    #pragma unroll
    for (int r = 0; r < 8; ++r) {
      const int m = m0 + r + 8 * halfsel;
      const int subj = subject[base + m];
      codes[m] = (subj == 0) ? bi1[r] : bi2[r];
    }
  }
  __syncthreads();

  for (int idx = tid; idx < 128 * 128; idx += 256) {
    const int m = idx >> 7, col = idx & 127;
    out_c[(base + m) * ZD + col] = Cws[(size_t)codes[m] * ZD + col];
  }
}

// ---------------------------------------------------------------------------
extern "C" void kernel_launch(void* const* d_in, const int* in_sizes, int n_in,
                              void* d_out, int out_size, void* d_ws, size_t ws_size,
                              hipStream_t stream) {
  (void)in_sizes; (void)n_in; (void)out_size; (void)ws_size;
  const float* z_e     = (const float*)d_in[0];
  const int*   subject = (const int*)  d_in[1];
  const float* e1      = (const float*)d_in[2];
  const float* e2      = (const float*)d_in[3];
  const float* W1      = (const float*)d_in[4];
  const float* b1      = (const float*)d_in[5];
  const float* W2      = (const float*)d_in[6];
  const float* b2      = (const float*)d_in[7];

  float* out_z = (float*)d_out;
  float* out_c = out_z + (size_t)NROW * ZD;

  char* ws = (char*)d_ws;
  float*    Cws   = (float*)ws;                         // 512 KB
  _Float16* Eh    = (_Float16*)(ws + 524288);           // 256 KB
  float*    enorm = (float*)(ws + 786432);              //   4 KB
  _Float16* W1T   = (_Float16*)(ws + 790528);           // 2,588,672 B
  _Float16* W2T   = (_Float16*)(ws + 3379200);          // 2,588,672 B

  KLCLR_prep  <<<128,  256, 0, stream>>>(e1, e2, Eh, enorm);
  KLCLR_wprep <<<1264, 256, 0, stream>>>(W1, W2, W1T, W2T);
  KLCLR_mlp   <<<16,   256, 96 * 1024, stream>>>(e1, e2, W1T, W2T, b1, b2, Cws);
  KLCLR_assign<<<128,  256, 0, stream>>>(z_e, subject, Eh, enorm, Cws, out_z, out_c);
}
